// JointAttention_4913442586932
// MI455X (gfx1250) — matrix operand order repaired and checked by reading, not compiled, not run on hardware
//
#include <hip/hip_runtime.h>
#include <hip/hip_bf16.h>

#define DIM   1536
#define NHEAD 24
#define HDIM  64
#define S_IMG 2048
#define T_TXT 512
#define L_TOT 2560

typedef __bf16 bf16_t;
typedef __attribute__((ext_vector_type(16))) __bf16 bf16x16;
typedef __attribute__((ext_vector_type(8)))  __bf16 bf16x8;
typedef __attribute__((ext_vector_type(8)))  float  f32x8;
typedef __attribute__((ext_vector_type(4)))  int    i32x4;

// ---- CDNA5 async Global->LDS path (guarded probes) ----
#if defined(__has_builtin)
#if __has_builtin(__builtin_amdgcn_global_load_async_to_lds_b128)
#define HAVE_ASYNC_LDS 1
#endif
#if __has_builtin(__builtin_amdgcn_s_wait_asynccnt)
#define HAVE_WAIT_ASYNC 1
#endif
#endif

typedef __attribute__((address_space(1))) i32x4* as1_v4i;   // global, typed per builtin sig
typedef __attribute__((address_space(3))) i32x4* as3_v4i;   // LDS

__device__ __forceinline__ void wait_async_le(int n) {
#if defined(HAVE_ASYNC_LDS)
#if defined(HAVE_WAIT_ASYNC)
    if (n == 0) __builtin_amdgcn_s_wait_asynccnt(0);
    else        __builtin_amdgcn_s_wait_asynccnt(1);
#else
    if (n == 0) asm volatile("s_wait_asynccnt 0x0" ::: "memory");
    else        asm volatile("s_wait_asynccnt 0x1" ::: "memory");
#endif
#else
    (void)n;
#endif
}

// ---------------- helpers ----------------

__device__ __forceinline__ unsigned short f2bf(float f) {
    union { float f; unsigned int u; } v; v.f = f;
    unsigned int u = v.u;
    u += 0x7FFFu + ((u >> 16) & 1u);   // round-to-nearest-even
    return (unsigned short)(u >> 16);
}

union FragAB { bf16x16 v; bf16x8 h[2]; };

// A/B fragment from two 16B runs per lane
__device__ __forceinline__ bf16x16 load_frag_split(const bf16_t* lo, const bf16_t* hi) {
    FragAB f;
    f.h[0] = *(const bf16x8*)lo;
    f.h[1] = *(const bf16x8*)hi;
    return f.v;
}

// reduce across the 16 lanes that share one C-matrix row (lanes 0-15 / 16-31)
__device__ __forceinline__ float rowmax16(float x) {
    x = fmaxf(x, __shfl_xor(x, 1, 32));
    x = fmaxf(x, __shfl_xor(x, 2, 32));
    x = fmaxf(x, __shfl_xor(x, 4, 32));
    x = fmaxf(x, __shfl_xor(x, 8, 32));
    return x;
}
__device__ __forceinline__ float rowsum16(float x) {
    x += __shfl_xor(x, 1, 32);
    x += __shfl_xor(x, 2, 32);
    x += __shfl_xor(x, 4, 32);
    x += __shfl_xor(x, 8, 32);
    return x;
}

// ---------------- fp32 -> bf16 conversion ----------------

__global__ void cvt_bf16_kernel(const float* __restrict__ in,
                                unsigned short* __restrict__ out, int n) {
    int i = blockIdx.x * blockDim.x + threadIdx.x;
    if (i < n) out[i] = f2bf(in[i]);
}

// ---------------- GEMM:  Y = X @ W^T + b  (+ optional per-head LayerNorm) ----------------
// X: (M, DIM) bf16 row-major.  W: (DIM, DIM) bf16 row-major (out,in) == N x K.
// Block: 256 threads = 8 waves.  Wave computes 16 rows x 64 cols (one full head).
// W tile (64 x 32) is block-uniform -> staged through LDS with double-buffered
// async Global->LDS DMA (ASYNCcnt), padded rows (40 elems) for bank-conflict-free reads.
#define WPAD 40   // 32 k-elements + 8 pad -> 20-dword row stride, 16 distinct banks

template<bool LN, bool F32OUT>
__global__ void __launch_bounds__(256)
gemm_bias_ln(const unsigned short* __restrict__ Xr,
             const unsigned short* __restrict__ Wr,
             const float* __restrict__ bias,
             const float* __restrict__ gamma,
             void* __restrict__ outp)
{
    __shared__ unsigned short ldsW[2][64 * WPAD];

    const bf16_t* X = (const bf16_t*)Xr;

    const int tid  = threadIdx.x;
    const int lane = tid & 31;
    const int wave = tid >> 5;
    const int hi   = lane >> 4;
    const int ln16 = lane & 15;

    const int m0 = blockIdx.x * 128 + wave * 16;
    const int n0 = blockIdx.y * 64;

    // staging coordinates: each of 256 threads moves one 16B slice (8 bf16)
    const int sn = tid >> 2;          // 0..63  (row of W tile)
    const int sk = (tid & 3) * 8;     // 0,8,16,24 (k offset)

    auto stage = [&](int kk, int buf) {
        const unsigned short* gp = Wr + (size_t)(n0 + sn) * DIM + kk + sk;
        unsigned short* lp = &ldsW[buf][sn * WPAD + sk];
#if defined(HAVE_ASYNC_LDS)
        __builtin_amdgcn_global_load_async_to_lds_b128((as1_v4i)gp, (as3_v4i)lp, 0, 0);
#else
        *(bf16x8*)lp = *(const bf16x8*)gp;
#endif
    };

    f32x8 acc[4] = { {}, {}, {}, {} };

    const bf16_t* xrow = X + (size_t)(m0 + ln16) * DIM;

    const int NK = DIM / 32;          // 48 k-steps
    stage(0, 0);

    for (int ki = 0; ki < NK; ++ki) {
        const int kk  = ki * 32;
        const int buf = ki & 1;
        if (ki + 1 < NK) {
            stage(kk + 32, buf ^ 1);
            wait_async_le(1);         // oldest (current buf) complete
        } else {
            wait_async_le(0);
        }
        __syncthreads();

        if (kk + 32 < DIM) __builtin_prefetch(xrow + kk + 32, 0, 1);
        bf16x16 a = load_frag_split(xrow + kk + 8 * hi,
                                    xrow + kk + 16 + 8 * hi);
#pragma unroll
        for (int t = 0; t < 4; ++t) {
            const unsigned short* wb = &ldsW[buf][(t * 16 + ln16) * WPAD + 8 * hi];
            bf16x16 b = load_frag_split((const bf16_t*)wb, (const bf16_t*)wb + 16);
            acc[t] = __builtin_amdgcn_wmma_f32_16x16x32_bf16(
                false, a, false, b, (short)0, acc[t], false, false);
        }
        __syncthreads();              // protect buf before DMA overwrites it
    }

    float bsv[4], gv[4];
#pragma unroll
    for (int t = 0; t < 4; ++t) {
        const int col = n0 + t * 16 + ln16;
        bsv[t] = bias[col];
        if (LN) gv[t] = gamma[t * 16 + ln16];   // head-local index
    }
#pragma unroll
    for (int t = 0; t < 4; ++t)
#pragma unroll
        for (int r = 0; r < 8; ++r)
            acc[t][r] += bsv[t];

    if (LN) {
#pragma unroll
        for (int r = 0; r < 8; ++r) {
            float s = 0.f, s2 = 0.f;
#pragma unroll
            for (int t = 0; t < 4; ++t) { float v = acc[t][r]; s += v; s2 += v * v; }
            s  = rowsum16(s);
            s2 = rowsum16(s2);
            const float mean = s * (1.0f / 64.0f);
            const float var  = s2 * (1.0f / 64.0f) - mean * mean;
            const float rstd = rsqrtf(var + 1e-5f);
#pragma unroll
            for (int t = 0; t < 4; ++t)
                acc[t][r] = (acc[t][r] - mean) * rstd * gv[t];
        }
    }

#pragma unroll
    for (int r = 0; r < 8; ++r) {
        const int row = m0 + r + 8 * hi;
#pragma unroll
        for (int t = 0; t < 4; ++t) {
            const int col = n0 + t * 16 + ln16;
            if (F32OUT)
                ((float*)outp)[(size_t)row * DIM + col] = acc[t][r];
            else
                ((unsigned short*)outp)[(size_t)row * DIM + col] = f2bf(acc[t][r]);
        }
    }
}

// ---------------- Flash joint attention ----------------
// Grid: (L/64, NHEAD). Block: 128 threads = 4 waves; wave = 16 queries x 64 dim.
__global__ void __launch_bounds__(128)
joint_attention(const unsigned short* __restrict__ Qr,
                const unsigned short* __restrict__ Kr,
                const unsigned short* __restrict__ Vr,
                unsigned short* __restrict__ Or)
{
    __shared__ unsigned short ldsV[64 * 64];      // V^T tile: [d][key]
    __shared__ unsigned short ldsP[4 * 16 * 64];  // P tiles per wave: [q][key]

    const bf16_t* Q = (const bf16_t*)Qr;
    const bf16_t* K = (const bf16_t*)Kr;
    const bf16_t* V = (const bf16_t*)Vr;

    const int tid  = threadIdx.x;
    const int lane = tid & 31;
    const int wave = tid >> 5;
    const int hi   = lane >> 4;
    const int ln16 = lane & 15;

    const int h  = blockIdx.y;
    const int q0 = blockIdx.x * 64 + wave * 16;
    const int hc = h * HDIM;

    // Q fragments (d chunks 0..31, 32..63), held for entire kernel
    const bf16_t* qrow = Q + (size_t)(q0 + ln16) * DIM + hc;
    bf16x16 qa[2];
#pragma unroll
    for (int j = 0; j < 2; ++j)
        qa[j] = load_frag_split(qrow + j * 32 + 16 * hi,
                                qrow + j * 32 + 16 * hi + 8);

    f32x8 o[4] = { {}, {}, {}, {} };
    float mrun[8], lrun[8];
#pragma unroll
    for (int r = 0; r < 8; ++r) { mrun[r] = -1e30f; lrun[r] = 0.f; }

    for (int kt = 0; kt < L_TOT / 64; ++kt) {
        // ---- stage V^T into LDS (coalesced read along d, transposed scatter) ----
#pragma unroll
        for (int r = 0; r < 4; ++r) {
            const int lv  = tid + r * 128;      // 0..511
            const int key = lv >> 3;
            const int db  = (lv & 7) * 8;
            const bf16_t* vp = V + (size_t)(kt * 64 + key) * DIM + hc + db;
            bf16x8 vv = *(const bf16x8*)vp;
            const unsigned short* vu = (const unsigned short*)&vv;
#pragma unroll
            for (int i = 0; i < 8; ++i)
                ldsV[(db + i) * 64 + key] = vu[i];
        }
        __syncthreads();

        // ---- S = (Q K^T) * scale ----
        f32x8 sc[4];
#pragma unroll
        for (int n = 0; n < 4; ++n) {
            f32x8 s = {};
#pragma unroll
            for (int j = 0; j < 2; ++j) {
                const bf16_t* krow =
                    K + (size_t)(kt * 64 + n * 16 + ln16) * DIM + hc + j * 32 + 16 * hi;
                bf16x16 b = *(const bf16x16*)krow;
                s = __builtin_amdgcn_wmma_f32_16x16x32_bf16(
                    false, qa[j], false, b, (short)0, s, false, false);
            }
#pragma unroll
            for (int r = 0; r < 8; ++r) s[r] *= 0.125f;   // 1/sqrt(64)
            sc[n] = s;
        }

        // ---- online softmax ----
        float mnew[8], alpha[8];
#pragma unroll
        for (int r = 0; r < 8; ++r) {
            float loc = sc[0][r];
            loc = fmaxf(loc, sc[1][r]);
            loc = fmaxf(loc, sc[2][r]);
            loc = fmaxf(loc, sc[3][r]);
            loc = rowmax16(loc);
            mnew[r]  = fmaxf(mrun[r], loc);
            alpha[r] = __expf(mrun[r] - mnew[r]);
        }
#pragma unroll
        for (int n = 0; n < 4; ++n)
#pragma unroll
            for (int r = 0; r < 8; ++r)
                sc[n][r] = __expf(sc[n][r] - mnew[r]);
#pragma unroll
        for (int r = 0; r < 8; ++r) {
            float ladd = sc[0][r] + sc[1][r] + sc[2][r] + sc[3][r];
            ladd = rowsum16(ladd);
            lrun[r] = lrun[r] * alpha[r] + ladd;
            mrun[r] = mnew[r];
        }
#pragma unroll
        for (int t = 0; t < 4; ++t)
#pragma unroll
            for (int r = 0; r < 8; ++r)
                o[t][r] *= alpha[r];

        // ---- stage P (C-layout -> row-major bf16) into per-wave LDS region ----
        unsigned short* pbase = ldsP + wave * 1024;
#pragma unroll
        for (int n = 0; n < 4; ++n)
#pragma unroll
            for (int r = 0; r < 8; ++r)
                pbase[(r + 8 * hi) * 64 + n * 16 + ln16] = f2bf(sc[n][r]);
        asm volatile("s_wait_dscnt 0x0" ::: "memory");   // per-wave LDS RAW fence

        // ---- O += P @ V ----
#pragma unroll
        for (int j = 0; j < 2; ++j) {
            const bf16_t* prow = (const bf16_t*)ldsP + wave * 1024 + ln16 * 64 + j * 32;
            bf16x16 pa = load_frag_split(prow + 16 * hi, prow + 16 * hi + 8);
#pragma unroll
            for (int t = 0; t < 4; ++t) {
                const bf16_t* vrow =
                    (const bf16_t*)ldsV + (t * 16 + ln16) * 64 + j * 32 + 16 * hi;
                bf16x16 vb = *(const bf16x16*)vrow;
                o[t] = __builtin_amdgcn_wmma_f32_16x16x32_bf16(
                    false, pa, false, vb, (short)0, o[t], false, false);
            }
        }
        __syncthreads();
    }

    // ---- normalize and store ----
#pragma unroll
    for (int r = 0; r < 8; ++r) {
        const float inv = 1.0f / lrun[r];
        const int row = q0 + r + 8 * hi;
#pragma unroll
        for (int t = 0; t < 4; ++t)
            Or[(size_t)row * DIM + hc + t * 16 + ln16] = f2bf(o[t][r] * inv);
    }
}

// ---------------- launcher ----------------

extern "C" void kernel_launch(void* const* d_in, const int* in_sizes, int n_in,
                              void* d_out, int out_size, void* d_ws, size_t ws_size,
                              hipStream_t stream)
{
    const float* hs  = (const float*)d_in[0];
    const float* ehs = (const float*)d_in[1];
    const float* Wq  = (const float*)d_in[2];  const float* bq  = (const float*)d_in[3];
    const float* Wk  = (const float*)d_in[4];  const float* bk  = (const float*)d_in[5];
    const float* Wv  = (const float*)d_in[6];  const float* bv  = (const float*)d_in[7];
    const float* Waq = (const float*)d_in[8];  const float* baq = (const float*)d_in[9];
    const float* Wak = (const float*)d_in[10]; const float* bak = (const float*)d_in[11];
    const float* Wav = (const float*)d_in[12]; const float* bav = (const float*)d_in[13];
    const float* Wo  = (const float*)d_in[14]; const float* bo  = (const float*)d_in[15];
    const float* Wao = (const float*)d_in[16]; const float* bao = (const float*)d_in[17];
    const float* gq  = (const float*)d_in[18]; const float* gk  = (const float*)d_in[19];
    const float* gaq = (const float*)d_in[20]; const float* gak = (const float*)d_in[21];

    const size_t NMAT = (size_t)DIM * DIM;     // one weight matrix
    const size_t NL   = (size_t)L_TOT * DIM;   // one (L, DIM) activation

    unsigned short* Xc = (unsigned short*)d_ws;          // bf16 concat input
    unsigned short* Qc = Xc + NL;
    unsigned short* Kc = Qc + NL;
    unsigned short* Vc = Kc + NL;
    unsigned short* AO = Vc + NL;                        // attention output (bf16)
    unsigned short* Wc0 = AO + NL;                       // 8 bf16 weight matrices
    unsigned short* Wcs[8];
    for (int i = 0; i < 8; ++i) Wcs[i] = Wc0 + (size_t)i * NMAT;

    const int TPB = 256;
    auto cvt = [&](const float* src, unsigned short* dst, size_t n) {
        cvt_bf16_kernel<<<dim3((unsigned)((n + TPB - 1) / TPB)), dim3(TPB), 0, stream>>>(
            src, dst, (int)n);
    };

    cvt(hs,  Xc,                         (size_t)S_IMG * DIM);
    cvt(ehs, Xc + (size_t)S_IMG * DIM,   (size_t)T_TXT * DIM);
    const float* Ws[8] = { Wq, Wk, Wv, Waq, Wak, Wav, Wo, Wao };
    for (int i = 0; i < 8; ++i) cvt(Ws[i], Wcs[i], NMAT);

    const dim3 blk(256);
    const dim3 gi(S_IMG / 128, DIM / 64);   // (16, 24)
    const dim3 gt(T_TXT / 128, DIM / 64);   // (4, 24)
    const size_t offT = (size_t)S_IMG * DIM;

    // projections (img stream: Wq/Wk/Wv ; txt stream: Waq/Wak/Wav)
    gemm_bias_ln<true,  false><<<gi, blk, 0, stream>>>(Xc,        Wcs[0], bq,  gq,      Qc);
    gemm_bias_ln<true,  false><<<gi, blk, 0, stream>>>(Xc,        Wcs[1], bk,  gk,      Kc);
    gemm_bias_ln<false, false><<<gi, blk, 0, stream>>>(Xc,        Wcs[2], bv,  nullptr, Vc);
    gemm_bias_ln<true,  false><<<gt, blk, 0, stream>>>(Xc + offT, Wcs[3], baq, gaq,     Qc + offT);
    gemm_bias_ln<true,  false><<<gt, blk, 0, stream>>>(Xc + offT, Wcs[4], bak, gak,     Kc + offT);
    gemm_bias_ln<false, false><<<gt, blk, 0, stream>>>(Xc + offT, Wcs[5], bav, nullptr, Vc + offT);

    // joint attention over concatenated 2560 tokens
    joint_attention<<<dim3(L_TOT / 64, NHEAD), dim3(128), 0, stream>>>(Qc, Kc, Vc, AO);

    // output projections straight to fp32 d_out (img then txt, flat-concatenated)
    gemm_bias_ln<false, true><<<gi, blk, 0, stream>>>(AO,        Wcs[6], bo,  nullptr, d_out);
    gemm_bias_ln<false, true><<<gt, blk, 0, stream>>>(AO + offT, Wcs[7], bao, nullptr,
                                                      (float*)d_out + offT);
}
